// ScaledDotProdAttn_26302379721280
// MI455X (gfx1250) — matrix operand order repaired and checked by reading, not compiled, not run on hardware
//
#include <hip/hip_runtime.h>

typedef __attribute__((ext_vector_type(16))) _Float16 v16h;
typedef __attribute__((ext_vector_type(8)))  _Float16 v8h;
typedef __attribute__((ext_vector_type(4)))  _Float16 v4h;
typedef __attribute__((ext_vector_type(8)))  float    v8f;

#define DIMH    64
#define SEQ     2048
#define BATCH   32
#define BLOCK_M 128
#define BLOCK_N 64
#define NWAVE   8
#define NCHUNK  (SEQ / BLOCK_N)

__device__ __forceinline__ v16h mk16(v8h a, v8h b) {
  v16h r;
#pragma unroll
  for (int i = 0; i < 8; ++i) { r[i] = a[i]; r[i + 8] = b[i]; }
  return r;
}

// Load one 64x64 f32 chunk of K and V into registers (4 float4 each / thread).
__device__ __forceinline__ void load_chunk(const float* kb, const float* vb,
                                           int tid, float4* kr, float4* vr) {
#pragma unroll
  for (int i = 0; i < 4; ++i) {
    const int linear = (i * 256 + tid) * 4;
    kr[i] = *(const float4*)(kb + linear);
    vr[i] = *(const float4*)(vb + linear);
  }
}

// Convert staged registers to f16 and store: K row-major, V transposed.
__device__ __forceinline__ void stage_to_lds(const float4* kr, const float4* vr,
                                             _Float16* kl, _Float16* vl, int tid) {
#pragma unroll
  for (int i = 0; i < 4; ++i) {
    const int linear = (i * 256 + tid) * 4;
    const int key = linear >> 6;
    const int dim = linear & 63;
    v4h kh; kh[0] = (_Float16)kr[i].x; kh[1] = (_Float16)kr[i].y;
            kh[2] = (_Float16)kr[i].z; kh[3] = (_Float16)kr[i].w;
    *(v4h*)(&kl[linear]) = kh;
    vl[(dim + 0) * BLOCK_N + key] = (_Float16)vr[i].x;
    vl[(dim + 1) * BLOCK_N + key] = (_Float16)vr[i].y;
    vl[(dim + 2) * BLOCK_N + key] = (_Float16)vr[i].z;
    vl[(dim + 3) * BLOCK_N + key] = (_Float16)vr[i].w;
  }
}

// Flash attention: out = softmax(Q K^T / sqrt(D)) V. One 128x64 Q tile per WG,
// each wave32 owns 16 query rows; double-buffered K/V tiles in LDS; all
// matmuls via v_wmma_f32_16x16x32_f16 (f32 accumulate).
__global__ __launch_bounds__(256) void fa_kernel(const float* __restrict__ q,
                                                 const float* __restrict__ k,
                                                 const float* __restrict__ v,
                                                 float* __restrict__ out) {
  __shared__ _Float16 k_lds[2][BLOCK_N * DIMH];    // [buf][key][dim]
  __shared__ _Float16 vt_lds[2][DIMH * BLOCK_N];   // [buf][dim][key]
  __shared__ _Float16 p_lds[NWAVE * 16 * DIMH];    // per-wave P staging [row][key]

  const int tid   = threadIdx.x;
  const int wave  = tid >> 5;
  const int lane  = tid & 31;
  const int laneM = lane & 15;   // row (A) / col (B,C) index
  const int hi    = lane >> 4;   // half-wave select

  const int b  = blockIdx.x / (SEQ / BLOCK_M);
  const int mt = blockIdx.x % (SEQ / BLOCK_M);
  const int q0 = mt * BLOCK_M + wave * 16;         // first query row of this wave

  // ---- Load Q A-fragments (16x32 f16, two K-steps), fold in 1/sqrt(64) ----
  // A layout: lane holds row M=laneM; elems 0..7 at K = 32f + 8*hi + e,
  // elems 8..15 at K = 32f + 16 + 8*hi + (e-8)  (two contiguous groups).
  const float* qrow = q + ((size_t)b * SEQ + q0 + laneM) * DIMH;
  v16h qa[2];
#pragma unroll
  for (int f = 0; f < 2; ++f) {
    const int g0 = 32 * f + 8 * hi;
    const float4 x0 = *(const float4*)(qrow + g0);
    const float4 x1 = *(const float4*)(qrow + g0 + 4);
    const float4 y0 = *(const float4*)(qrow + g0 + 16);
    const float4 y1 = *(const float4*)(qrow + g0 + 20);
    const float gx[8] = {x0.x, x0.y, x0.z, x0.w, x1.x, x1.y, x1.z, x1.w};
    const float gy[8] = {y0.x, y0.y, y0.z, y0.w, y1.x, y1.y, y1.z, y1.w};
#pragma unroll
    for (int i = 0; i < 8; ++i) {
      qa[f][i]     = (_Float16)(gx[i] * 0.125f);   // 1/sqrt(64)
      qa[f][i + 8] = (_Float16)(gy[i] * 0.125f);
    }
  }

  // ---- Online-softmax state: rows 8*hi + r, r = 0..7 (matches C layout) ----
  v8f o[4];
#pragma unroll
  for (int d = 0; d < 4; ++d)
#pragma unroll
    for (int r = 0; r < 8; ++r) o[d][r] = 0.0f;
  float m_r[8], l_r[8];
#pragma unroll
  for (int r = 0; r < 8; ++r) { m_r[r] = -1e30f; l_r[r] = 0.0f; }

  _Float16* pw = &p_lds[wave * 16 * DIMH];
  const float* kbase = k + (size_t)b * SEQ * DIMH;
  const float* vbase = v + (size_t)b * SEQ * DIMH;

  // ---- Prologue: stage chunk 0 into buffer 0 ----
  float4 kreg[4], vreg[4];
  load_chunk(kbase, vbase, tid, kreg, vreg);
  stage_to_lds(kreg, vreg, k_lds[0], vt_lds[0], tid);

  for (int c = 0; c < NCHUNK; ++c) {
    __syncthreads();  // buffer c&1 fully staged; buffer (c+1)&1 free to write

    // Kick off next chunk's global loads now — latency hides behind compute.
    if (c + 1 < NCHUNK)
      load_chunk(kbase + (size_t)(c + 1) * BLOCK_N * DIMH,
                 vbase + (size_t)(c + 1) * BLOCK_N * DIMH, tid, kreg, vreg);

    const _Float16* kl = k_lds[c & 1];
    const _Float16* vl = vt_lds[c & 1];

    // ---- S = (Q/sqrtD) K^T : four 16x16 N-tiles, two K-steps each ----
    v8f s[4];
#pragma unroll
    for (int j = 0; j < 4; ++j) {
#pragma unroll
      for (int r = 0; r < 8; ++r) s[j][r] = 0.0f;
#pragma unroll
      for (int f = 0; f < 2; ++f) {
        // B layout: lane holds col N=laneM (key j*16+laneM), K = 32f+8*hi+e, then 32f+16+8*hi+(e-8)
        const _Float16* bp = &kl[(j * 16 + laneM) * DIMH + 32 * f + 8 * hi];
        const v16h kB = mk16(*(const v8h*)bp, *(const v8h*)(bp + 16));
        s[j] = __builtin_amdgcn_wmma_f32_16x16x32_f16(
            false, qa[f], false, kB, (short)0, s[j], false, false);
      }
    }

    // ---- online softmax over this 64-key chunk ----
    float mc[8];
#pragma unroll
    for (int r = 0; r < 8; ++r)
      mc[r] = fmaxf(fmaxf(s[0][r], s[1][r]), fmaxf(s[2][r], s[3][r]));
#pragma unroll
    for (int mask = 1; mask <= 8; mask <<= 1)
#pragma unroll
      for (int r = 0; r < 8; ++r)
        mc[r] = fmaxf(mc[r], __shfl_xor(mc[r], mask, 32));  // stays in 16-lane half

    float scale[8];
#pragma unroll
    for (int r = 0; r < 8; ++r) {
      const float mn = fmaxf(m_r[r], mc[r]);
      scale[r] = __expf(m_r[r] - mn);
      m_r[r] = mn;
    }

    float rs[8];
#pragma unroll
    for (int r = 0; r < 8; ++r) rs[r] = 0.0f;
#pragma unroll
    for (int j = 0; j < 4; ++j)
#pragma unroll
      for (int r = 0; r < 8; ++r) {
        const float p = __expf(s[j][r] - m_r[r]);
        rs[r] += p;
        pw[(hi * 8 + r) * DIMH + j * 16 + laneM] = (_Float16)p;  // wave-private
      }
#pragma unroll
    for (int mask = 1; mask <= 8; mask <<= 1)
#pragma unroll
      for (int r = 0; r < 8; ++r) rs[r] += __shfl_xor(rs[r], mask, 32);
#pragma unroll
    for (int r = 0; r < 8; ++r) l_r[r] = l_r[r] * scale[r] + rs[r];
#pragma unroll
    for (int d = 0; d < 4; ++d)
#pragma unroll
      for (int r = 0; r < 8; ++r) o[d][r] *= scale[r];

    // ---- O += P V : re-load P in A layout from LDS (C->A transpose) ----
#pragma unroll
    for (int f = 0; f < 2; ++f) {
      const _Float16* ap = &pw[laneM * DIMH + 32 * f + 8 * hi];
      const v16h pA = mk16(*(const v8h*)ap, *(const v8h*)(ap + 16));
#pragma unroll
      for (int d = 0; d < 4; ++d) {
        const _Float16* bp = &vl[(d * 16 + laneM) * BLOCK_N + 32 * f + 8 * hi];
        const v16h vB = mk16(*(const v8h*)bp, *(const v8h*)(bp + 16));
        o[d] = __builtin_amdgcn_wmma_f32_16x16x32_f16(
            false, pA, false, vB, (short)0, o[d], false, false);
      }
    }

    // ---- Stage next chunk into the other buffer (visible after next barrier) ----
    if (c + 1 < NCHUNK)
      stage_to_lds(kreg, vreg, k_lds[(c + 1) & 1], vt_lds[(c + 1) & 1], tid);
  }

  // ---- epilogue: O / l, scatter per C layout (row = 8*hi + r, col = laneM) ----
  float* ob = out + ((size_t)b * SEQ + q0) * DIMH;
#pragma unroll
  for (int r = 0; r < 8; ++r) {
    const float inv = 1.0f / l_r[r];
    float* orow = ob + (hi * 8 + r) * DIMH + laneM;
#pragma unroll
    for (int d = 0; d < 4; ++d) orow[d * 16] = o[d][r] * inv;
  }
}

extern "C" void kernel_launch(void* const* d_in, const int* in_sizes, int n_in,
                              void* d_out, int out_size, void* d_ws, size_t ws_size,
                              hipStream_t stream) {
  (void)in_sizes; (void)n_in; (void)out_size; (void)d_ws; (void)ws_size;
  const float* q = (const float*)d_in[0];
  const float* k = (const float*)d_in[1];
  const float* v = (const float*)d_in[2];
  float* out = (float*)d_out;
  const int grid = BATCH * (SEQ / BLOCK_M);  // 32 * 16 = 512 workgroups
  fa_kernel<<<grid, 256, 0, stream>>>(q, k, v, out);
}
